// CompositionNet_82875688944112
// MI455X (gfx1250) — matrix operand order repaired and checked by reading, not compiled, not run on hardware
//
#include <hip/hip_runtime.h>

// CompositionNet (Roost-style) for MI455X gfx1250, wave32 + WMMA f16->f32.
// Round 4: generic GEMM software-pipelined with double-buffered LDS tiles
// (async B copy + batched float4 A loads for tile t+1 issued before computing
// tile t; one barrier per k-tile). A-staging loads batched into registers so
// global_load_b128s issue back-to-back instead of load/wait/convert chains.

#define NCRY   8192
#define NATOM  65536
#define NEDGE  524288

typedef __attribute__((ext_vector_type(16))) _Float16 v16h;
typedef __attribute__((ext_vector_type(4)))  _Float16 v4h;
typedef __attribute__((ext_vector_type(8)))  float    v8f;

#define WMMA_F16(af, bf, c) \
  __builtin_amdgcn_wmma_f32_16x16x32_f16(false, (af), false, (bf), (short)0, (c), false, false)

// ---------- A fragment (16x32 f16) from LDS row-major (ISA 7.12.2) ----------
__device__ __forceinline__ v16h frag_a(const _Float16* s, int r0, int k0, int ld, int lane) {
  v16h a;
  int row  = r0 + (lane & 15);
  int koff = k0 + ((lane < 16) ? 0 : 16);
  #pragma unroll
  for (int h = 0; h < 16; ++h) {
    int k = h;                          // contiguous 16 halves -> 2x ds_load_b128
    a[h] = s[row * ld + koff + k];
  }
  return a;
}

// ---------- B fragment from pre-swizzled layout: one aligned 32B read -------
// fragment (kt, ct) = 512 halves at ((kt*nct + ct)*512), order lane*16 + h,
// with k = kt*32 + (lane>=16?16:0) + h, n = ct*16 + (lane&15).
__device__ __forceinline__ v16h frag_b_swz(const _Float16* s, int kt, int ct, int nct, int lane) {
  return *(const v16h*)(s + (((kt * nct + ct) << 9) + (lane << 4)));
}

// ---------- CDNA5 async global->LDS copy (ASYNCcnt) -------------------------
__device__ __forceinline__ void async_copy_b128(void* ldsDst, const void* gSrc) {
  unsigned loff = (unsigned)(unsigned long long)ldsDst;     // low 32b = LDS offset
  unsigned long long ga = (unsigned long long)gSrc;
  asm volatile("global_load_async_to_lds_b128 %0, %1, off"
               :: "v"(loff), "v"(ga) : "memory");
}
__device__ __forceinline__ void wait_async() {
  asm volatile("s_wait_asynccnt 0x0" ::: "memory");
}

// Stage swizzled fragments [kt0..kt0+nkt) x [ct0..ct0+nct), NT = total col-tiles.
__device__ __forceinline__ void stage_wswz(_Float16* sW, const _Float16* src,
                                           int NT, int kt0, int ct0,
                                           int nkt, int nct, int tid) {
  const int chunk = nct * 512;
  for (int kt = 0; kt < nkt; ++kt) {
    const _Float16* g = src + (((size_t)(kt0 + kt) * NT + ct0) << 9);
    _Float16* d = sW + kt * chunk;
    for (int j = tid * 8; j < chunk; j += 2048)
      async_copy_b128(d + j, g + j);
  }
}

// Per-wave GEMM over its 16 rows with swizzled weights in LDS.
__device__ __forceinline__ void wave_gemm(const _Float16* sA, int lda,
                                          const _Float16* sW, int nct, int kdim,
                                          const float* bias, int relu,
                                          _Float16* sOut, int ldo, int wr, int lane) {
  for (int ct = 0; ct < nct; ++ct) {
    v8f acc = {};
    for (int kt = 0; kt < (kdim >> 5); ++kt)
      acc = WMMA_F16(frag_a(sA, wr, kt * 32, lda, lane),
                     frag_b_swz(sW, kt, ct, nct, lane), acc);
    int col = ct * 16 + (lane & 15);
    float bv = bias[col];
    #pragma unroll
    for (int g = 0; g < 8; ++g) {
      int row = wr + g + ((lane < 16) ? 0 : 8);
      float v = acc[g] + bv;
      if (relu) v = fmaxf(v, 0.f);
      sOut[row * ldo + col] = (_Float16)v;
    }
  }
}

// ---------------- generic tiled WMMA GEMM: C = act(A_f32 @ Bswz + bias)+addC
// Software-pipelined: double-buffered LDS tiles, stage t+1 while computing t.
__global__ __launch_bounds__(256)
void wmma_gemm(const float* __restrict__ A, const _Float16* __restrict__ B,
               const float* __restrict__ bias, const float* __restrict__ addC,
               float* __restrict__ C, int M, int N, int K, int Kp, int relu) {
  __shared__ __attribute__((aligned(32))) _Float16 sA[2][128 * 32];
  __shared__ __attribute__((aligned(32))) _Float16 sB[2][4 * 512];
  const int tid = threadIdx.x, lane = tid & 31, wave = tid >> 5;
  const int m0 = blockIdx.x * 128, n0 = blockIdx.y * 64;
  const int wr = (wave & 3) * 32, wc = (wave >> 2) * 32;
  const int NT = N >> 4, ct0 = n0 >> 4;
  const int ntiles = Kp >> 5;

  auto stage = [&](int buf, int k0) {
    // async B: 4 fragments, 2048 contiguous halves, 1 b128 per thread
    async_copy_b128(sB[buf] + tid * 8,
                    B + (((size_t)(k0 >> 5) * NT + ct0) << 9) + tid * 8);
    if (k0 + 32 <= K) {
      // batched: 4 global_load_b128 issued back-to-back, then cvt + ds_store
      float4 f[4];
      #pragma unroll
      for (int u = 0; u < 4; ++u) {
        int i = tid * 4 + u * 1024;
        int r = i >> 5, c = i & 31;
        f[u] = *(const float4*)(A + (long long)(m0 + r) * K + k0 + c);
      }
      #pragma unroll
      for (int u = 0; u < 4; ++u) {
        int i = tid * 4 + u * 1024;
        v4h hv; hv[0] = (_Float16)f[u].x; hv[1] = (_Float16)f[u].y;
                hv[2] = (_Float16)f[u].z; hv[3] = (_Float16)f[u].w;
        *(v4h*)(sA[buf] + i) = hv;
      }
    } else {
      // partial K-tile (embedding K=200 last tile only): guarded scalar path
      for (int i = tid; i < 128 * 32; i += 256) {
        int r = i >> 5, c = i & 31, gc = k0 + c;
        sA[buf][i] = (gc < K) ? (_Float16)A[(long long)(m0 + r) * K + gc]
                              : (_Float16)0.f;
      }
    }
  };

  v8f acc[2][2] = {};
  stage(0, 0);
  for (int t = 0; t < ntiles; ++t) {
    wait_async();
    __syncthreads();                       // tile t fully staged, prev reads done
    if (t + 1 < ntiles) stage((t + 1) & 1, (t + 1) * 32);
    const _Float16* cA = sA[t & 1];
    const _Float16* cB = sB[t & 1];
    v16h a0f = frag_a(cA, wr,      0, 32, lane);
    v16h a1f = frag_a(cA, wr + 16, 0, 32, lane);
    v16h b0f = frag_b_swz(cB, 0, (wc >> 4),     4, lane);
    v16h b1f = frag_b_swz(cB, 0, (wc >> 4) + 1, 4, lane);
    acc[0][0] = WMMA_F16(a0f, b0f, acc[0][0]);
    acc[0][1] = WMMA_F16(a0f, b1f, acc[0][1]);
    acc[1][0] = WMMA_F16(a1f, b0f, acc[1][0]);
    acc[1][1] = WMMA_F16(a1f, b1f, acc[1][1]);
  }
  #pragma unroll
  for (int i = 0; i < 2; ++i)
    #pragma unroll
    for (int j = 0; j < 2; ++j) {
      int col = n0 + wc + j * 16 + (lane & 15);
      #pragma unroll
      for (int g = 0; g < 8; ++g) {
        int row = m0 + wr + i * 16 + g + ((lane < 16) ? 0 : 8);
        float v = acc[i][j][g];
        if (bias) v += bias[col];
        if (relu) v = fmaxf(v, 0.f);
        if (addC) v += addC[(long long)row * N + col];
        C[(long long)row * N + col] = v;
      }
    }
}

// ---------------- fused graph layer: 2 crystals (16 atoms, 128 edges) / block
__global__ __launch_bounds__(256)
void graph_layer(const float* __restrict__ feaIn, float* __restrict__ feaOut,
                 const _Float16* __restrict__ W1, const _Float16* __restrict__ Wo,
                 const _Float16* __restrict__ P1, const _Float16* __restrict__ P2,
                 const float* __restrict__ b1, const float* __restrict__ bo,
                 const float* __restrict__ pb1, const float* __restrict__ pb2,
                 const float* __restrict__ pwo, const float* __restrict__ pbo,
                 const float* __restrict__ aw, const int* __restrict__ nbr) {
  __shared__ __attribute__((aligned(32))) _Float16 sE[128 * 128];
  __shared__ __attribute__((aligned(32))) _Float16 sW[128 * 128];
  __shared__ __attribute__((aligned(32))) _Float16 sH[128 * 128];
  __shared__ __attribute__((aligned(32))) _Float16 sMh[128 * 64];
  __shared__ __attribute__((aligned(32))) _Float16 sH2[128 * 64];
  __shared__ __attribute__((aligned(16))) float    sFeaF[16 * 64];
  __shared__ __attribute__((aligned(16))) _Float16 sFeaH[16 * 64];
  __shared__ float sGate[128];
  __shared__ int   sNbr[128];

  const int tid = threadIdx.x, lane = tid & 31, wave = tid >> 5;
  const int wr = wave * 16;                 // 8 waves x 16 rows = 128 edges
  const int a0 = blockIdx.x * 16;           // first atom of this block
  const long long e0 = (long long)a0 * 8;   // first edge

  {
    int i = tid * 4;                        // 1024 floats = 1 float4/thread
    float4 f = *(const float4*)(feaIn + (long long)a0 * 64 + i);
    *(float4*)(sFeaF + i) = f;
    v4h hv; hv[0] = (_Float16)f.x; hv[1] = (_Float16)f.y;
            hv[2] = (_Float16)f.z; hv[3] = (_Float16)f.w;
    *(v4h*)(sFeaH + i) = hv;
  }
  if (tid < 128) sNbr[tid] = nbr[e0 + tid] - a0;   // local nbr in [0,16)
  __syncthreads();

  for (int i = tid; i < 128 * 128; i += 256) {     // build E = [self | nbr]
    int r = i >> 7, c = i & 127;
    int src = (c < 64) ? ((r >> 3) * 64 + c) : (sNbr[r] * 64 + (c - 64));
    sE[i] = sFeaH[src];
  }

  // ---- message MLP: m = relu(E @ W1 + b1) @ Wo + bo, hidden split 2x128 cols
  v8f macc[4] = {};
  for (int h = 0; h < 2; ++h) {
    __syncthreads();
    stage_wswz(sW, W1, 16, 0, 8 * h, 4, 8, tid);   // W1 col-tiles [8h,8h+8)
    wait_async();
    __syncthreads();
    wave_gemm(sE, 128, sW, 8, 128, b1 + h * 128, 1, sH, 128, wr, lane);
    __syncthreads();
    stage_wswz(sW, Wo, 4, 4 * h, 0, 4, 4, tid);    // Wo kt [4h,4h+4)
    wait_async();
    __syncthreads();
    #pragma unroll
    for (int ct = 0; ct < 4; ++ct)
      #pragma unroll
      for (int kt = 0; kt < 4; ++kt)
        macc[ct] = WMMA_F16(frag_a(sH, wr, kt * 32, 128, lane),
                            frag_b_swz(sW, kt, ct, 4, lane), macc[ct]);
  }
  #pragma unroll
  for (int ct = 0; ct < 4; ++ct) {
    int col = ct * 16 + (lane & 15);
    float bv = bo[col];
    #pragma unroll
    for (int g = 0; g < 8; ++g) {
      int row = wr + g + ((lane < 16) ? 0 : 8);
      sMh[row * 64 + col] = (_Float16)(macc[ct][g] + bv);
    }
  }

  // ---- pool gate MLP: h2 = relu(relu(m@P1+pb1)@P2+pb2), hidden split 2x96
  v8f hacc[4] = {};
  for (int h = 0; h < 2; ++h) {
    __syncthreads();
    stage_wswz(sW, P1, 12, 0, 6 * h, 2, 6, tid);   // P1 col-tiles [6h,6h+6)
    wait_async();
    __syncthreads();
    wave_gemm(sMh, 64, sW, 6, 64, pb1 + h * 96, 1, sH, 96, wr, lane);
    __syncthreads();
    stage_wswz(sW, P2, 4, 3 * h, 0, 3, 4, tid);    // P2 kt [3h,3h+3)
    wait_async();
    __syncthreads();
    #pragma unroll
    for (int ct = 0; ct < 4; ++ct)
      #pragma unroll
      for (int kt = 0; kt < 3; ++kt)
        hacc[ct] = WMMA_F16(frag_a(sH, wr, kt * 32, 96, lane),
                            frag_b_swz(sW, kt, ct, 4, lane), hacc[ct]);
  }
  #pragma unroll
  for (int ct = 0; ct < 4; ++ct) {
    int col = ct * 16 + (lane & 15);
    float bv = pb2[col];
    #pragma unroll
    for (int g = 0; g < 8; ++g) {
      int row = wr + g + ((lane < 16) ? 0 : 8);
      sH2[row * 64 + col] = (_Float16)fmaxf(hacc[ct][g] + bv, 0.f);
    }
  }
  __syncthreads();

  if (tid < 128) {                         // gate = h2 . pwo + pbo
    float s = 0.f;
    #pragma unroll
    for (int d = 0; d < 64; ++d) s += (float)sH2[tid * 64 + d] * pwo[d];
    sGate[tid] = s + pbo[0];
  }
  __syncthreads();

  // ---- weighted softmax attention over each atom's 8 edges + residual add
  {
    int ia = tid >> 4;                     // atom 0..15 (16 threads/atom)
    int dl = (tid & 15) * 4;               // 4 dims per thread
    float gmax = -3.0e38f, coef[8];
    #pragma unroll
    for (int j = 0; j < 8; ++j) gmax = fmaxf(gmax, sGate[ia * 8 + j]);
    float denom = 0.f;
    #pragma unroll
    for (int j = 0; j < 8; ++j) {
      float w = aw[a0 + sNbr[ia * 8 + j]];
      float e = w * __expf(sGate[ia * 8 + j] - gmax);
      coef[j] = e; denom += e;
    }
    float inv = 1.f / (denom + 1e-13f);
    float acc[4] = {0.f, 0.f, 0.f, 0.f};
    #pragma unroll
    for (int j = 0; j < 8; ++j) {
      float cj = coef[j];
      #pragma unroll
      for (int q = 0; q < 4; ++q)
        acc[q] += cj * (float)sMh[(ia * 8 + j) * 64 + dl + q];
    }
    float4 o;
    o.x = acc[0] * inv + sFeaF[ia * 64 + dl + 0];
    o.y = acc[1] * inv + sFeaF[ia * 64 + dl + 1];
    o.z = acc[2] * inv + sFeaF[ia * 64 + dl + 2];
    o.w = acc[3] * inv + sFeaF[ia * 64 + dl + 3];
    *(float4*)(feaOut + (long long)(a0 + ia) * 64 + dl) = o;
  }
}

// ---------------- small helper kernels ----------------
// f32 -> f16 with zero K-padding, written directly in B-fragment swizzle order
__global__ __launch_bounds__(256)
void cvt_swz(const float* __restrict__ src, _Float16* __restrict__ dst,
             int K, int N, int Kp) {
  int i = blockIdx.x * 256 + threadIdx.x;
  int NT = N >> 4;
  int total = (Kp >> 5) * NT * 512;
  if (i >= total) return;
  int h    = i & 15;
  int lane = (i >> 4) & 31;
  int t    = i >> 9;
  int ct   = t % NT;
  int kt   = t / NT;
  int k = kt * 32 + ((lane >> 4) << 4) + h;
  int n = ct * 16 + (lane & 15);
  dst[i] = (k < K) ? (_Float16)src[(size_t)k * N + n] : (_Float16)0.f;
}

__global__ __launch_bounds__(256)
void dot64(const float* __restrict__ X, const float* __restrict__ w,
           const float* __restrict__ b, float* __restrict__ out, int rows) {
  int n = blockIdx.x * 256 + threadIdx.x;
  if (n >= rows) return;
  const float* x = X + (long long)n * 64;
  float s = 0.f;
  #pragma unroll
  for (int d = 0; d < 64; d += 4) {
    float4 xv = *(const float4*)(x + d);
    float4 wv = *(const float4*)(w + d);
    s += xv.x * wv.x + xv.y * wv.y + xv.z * wv.z + xv.w * wv.w;
  }
  out[n] = s + b[0];
}

__global__ __launch_bounds__(256)
void cry_attn(const float* __restrict__ fea, const float* __restrict__ gate,
              const float* __restrict__ aw, float* __restrict__ crys) {
  int t = blockIdx.x * 256 + threadIdx.x;
  int c = t >> 4, dl = (t & 15) * 4;
  if (c >= NCRY) return;
  int n0 = c * 8;
  float gmax = -3.0e38f, coef[8];
  #pragma unroll
  for (int j = 0; j < 8; ++j) gmax = fmaxf(gmax, gate[n0 + j]);
  float denom = 0.f;
  #pragma unroll
  for (int j = 0; j < 8; ++j) {
    float e = aw[n0 + j] * __expf(gate[n0 + j] - gmax);
    coef[j] = e; denom += e;
  }
  float inv = 1.f / (denom + 1e-13f);
  float4 acc = {0.f, 0.f, 0.f, 0.f};
  #pragma unroll
  for (int j = 0; j < 8; ++j) {
    float4 f = *(const float4*)(fea + (long long)(n0 + j) * 64 + dl);
    acc.x += coef[j] * f.x; acc.y += coef[j] * f.y;
    acc.z += coef[j] * f.z; acc.w += coef[j] * f.w;
  }
  acc.x *= inv; acc.y *= inv; acc.z *= inv; acc.w *= inv;
  *(float4*)(crys + (long long)c * 64 + dl) = acc;
}

__global__ __launch_bounds__(256)
void out_head(const float* __restrict__ X, const float* __restrict__ W,
              const float* __restrict__ b, float* __restrict__ out, int rows) {
  int t = blockIdx.x * 256 + threadIdx.x;
  if (t >= rows * 2) return;
  int n = t >> 1, o = t & 1;
  float s = 0.f;
  #pragma unroll
  for (int d = 0; d < 64; ++d) s += X[(long long)n * 64 + d] * W[d * 2 + o];
  out[t] = s + b[o];
}

// ---------------- host ----------------
extern "C" void kernel_launch(void* const* d_in, const int* in_sizes, int n_in,
                              void* d_out, int out_size, void* d_ws, size_t ws_size,
                              hipStream_t stream) {
  (void)in_sizes; (void)n_in; (void)out_size; (void)ws_size;
  const float* atom_w  = (const float*)d_in[0];
  const float* orig    = (const float*)d_in[1];
  const int*   nbr_idx = (const int*)d_in[3];
  const float* emb_W = (const float*)d_in[5];
  const float* emb_b = (const float*)d_in[6];
  const float* msg_W1 = (const float*)d_in[7];
  const float* msg_b1 = (const float*)d_in[8];
  const float* msg_Wo = (const float*)d_in[9];
  const float* msg_bo = (const float*)d_in[10];
  const float* pg_W1 = (const float*)d_in[11];
  const float* pg_b1 = (const float*)d_in[12];
  const float* pg_W2 = (const float*)d_in[13];
  const float* pg_b2 = (const float*)d_in[14];
  const float* pg_Wo = (const float*)d_in[15];
  const float* pg_bo = (const float*)d_in[16];
  const float* cry_W1 = (const float*)d_in[17];
  const float* cry_b1 = (const float*)d_in[18];
  const float* cry_W2 = (const float*)d_in[19];
  const float* cry_b2 = (const float*)d_in[20];
  const float* cry_W3 = (const float*)d_in[21];
  const float* cry_b3 = (const float*)d_in[22];
  const float* cry_Wo = (const float*)d_in[23];
  const float* cry_bo = (const float*)d_in[24];
  const float* fcW[4] = {(const float*)d_in[25], (const float*)d_in[26],
                         (const float*)d_in[27], (const float*)d_in[28]};
  const float* fcb[4] = {(const float*)d_in[29], (const float*)d_in[30],
                         (const float*)d_in[31], (const float*)d_in[32]};
  const float* rsW[4] = {(const float*)d_in[33], (const float*)d_in[34],
                         (const float*)d_in[35], (const float*)d_in[36]};
  const float* outW = (const float*)d_in[37];
  const float* outb = (const float*)d_in[38];
  float* out = (float*)d_out;

  // workspace bump allocator (~100 MB total)
  char* base = (char*)d_ws; size_t off = 0;
  auto allocH = [&](size_t nh) { _Float16* p = (_Float16*)(base + off);
                                 off += ((nh * 2 + 255) / 256) * 256; return p; };
  auto allocF = [&](size_t nf) { float* p = (float*)(base + off);
                                 off += ((nf * 4 + 255) / 256) * 256; return p; };

  int rd[5] = {64, 448, 320, 192, 64};
  _Float16* wEmb = allocH(224 * 64);
  _Float16* wM1  = allocH((size_t)3 * 128 * 256);
  _Float16* wMo  = allocH((size_t)3 * 256 * 64);
  _Float16* wP1  = allocH((size_t)3 * 64 * 192);
  _Float16* wP2  = allocH((size_t)3 * 192 * 64);
  _Float16* wC1  = allocH(64 * 320);
  _Float16* wC2  = allocH(320 * 192);
  _Float16* wC3  = allocH(192 * 64);
  _Float16* wFc[4]; _Float16* wRs[4];
  for (int i = 0; i < 4; ++i) {
    wFc[i] = allocH((size_t)rd[i] * rd[i + 1]);
    wRs[i] = allocH((size_t)rd[i] * rd[i + 1]);
  }
  float* feaA  = allocF((size_t)NATOM * 64);
  float* feaB  = allocF((size_t)NATOM * 64);
  float* cb1   = allocF((size_t)8192 * 320);
  float* cb2   = allocF((size_t)8192 * 192);
  float* cb3   = allocF((size_t)8192 * 64);
  float* gateN = allocF(NATOM);
  float* crys  = allocF((size_t)NCRY * 64);
  float* rtmp  = allocF((size_t)NCRY * 448);
  float* rxa   = allocF((size_t)NCRY * 448);
  float* rxb   = allocF((size_t)NCRY * 448);

  auto cvt = [&](const float* src, _Float16* dst, int K, int N, int Kp) {
    int n = (Kp >> 5) * (N >> 4) * 512;
    cvt_swz<<<(n + 255) / 256, 256, 0, stream>>>(src, dst, K, N, Kp);
  };
  cvt(emb_W, wEmb, 200, 64, 224);
  for (int l = 0; l < 3; ++l) {
    cvt(msg_W1 + (size_t)l * 128 * 256, wM1 + (size_t)l * 128 * 256, 128, 256, 128);
    cvt(msg_Wo + (size_t)l * 256 * 64,  wMo + (size_t)l * 256 * 64,  256, 64, 256);
    cvt(pg_W1  + (size_t)l * 64 * 192,  wP1 + (size_t)l * 64 * 192,  64, 192, 64);
    cvt(pg_W2  + (size_t)l * 192 * 64,  wP2 + (size_t)l * 192 * 64,  192, 64, 192);
  }
  cvt(cry_W1, wC1, 64, 320, 64);
  cvt(cry_W2, wC2, 320, 192, 320);
  cvt(cry_W3, wC3, 192, 64, 192);
  for (int i = 0; i < 4; ++i) {
    cvt(fcW[i], wFc[i], rd[i], rd[i + 1], rd[i]);
    cvt(rsW[i], wRs[i], rd[i], rd[i + 1], rd[i]);
  }

  // embedding: fea = orig(65536x200) @ emb_W + b   (Kp = 224)
  wmma_gemm<<<dim3(NATOM / 128, 1), 256, 0, stream>>>(
      orig, wEmb, emb_b, nullptr, feaA, NATOM, 64, 200, 224, 0);

  // 3 fused graph layers (ping-pong fea)
  const float* fin = feaA; float* fout = feaB;
  for (int l = 0; l < 3; ++l) {
    graph_layer<<<NCRY / 2, 256, 0, stream>>>(
        fin, fout,
        wM1 + (size_t)l * 128 * 256, wMo + (size_t)l * 256 * 64,
        wP1 + (size_t)l * 64 * 192,  wP2 + (size_t)l * 192 * 64,
        msg_b1 + l * 256, msg_bo + l * 64, pg_b1 + l * 192, pg_b2 + l * 64,
        pg_Wo + l * 64, pg_bo + l, atom_w, nbr_idx);
    float* t = fout; fout = (float*)fin; fin = t;
  }
  const float* fea = fin;

  // crystal gate MLP, chunked in 8 x 8192-row slabs to bound workspace
  for (int c8 = 0; c8 < 8; ++c8) {
    const float* fc = fea + (size_t)c8 * 8192 * 64;
    wmma_gemm<<<dim3(64, 5), 256, 0, stream>>>(fc,  wC1, cry_b1, nullptr, cb1, 8192, 320, 64, 64, 1);
    wmma_gemm<<<dim3(64, 3), 256, 0, stream>>>(cb1, wC2, cry_b2, nullptr, cb2, 8192, 192, 320, 320, 1);
    wmma_gemm<<<dim3(64, 1), 256, 0, stream>>>(cb2, wC3, cry_b3, nullptr, cb3, 8192, 64, 192, 192, 1);
    dot64<<<8192 / 256, 256, 0, stream>>>(cb3, cry_Wo, cry_bo, gateN + (size_t)c8 * 8192, 8192);
  }

  cry_attn<<<NCRY / 16, 256, 0, stream>>>(fea, gateN, atom_w, crys);

  // residual head: x = relu(x@W+b) + x@Wr, 4 stages
  const float* x = crys;
  float* ping[2] = {rxa, rxb};
  for (int i = 0; i < 4; ++i) {
    wmma_gemm<<<dim3(NCRY / 128, rd[i + 1] / 64), 256, 0, stream>>>(
        x, wFc[i], fcb[i], nullptr, rtmp, NCRY, rd[i + 1], rd[i], rd[i], 1);
    wmma_gemm<<<dim3(NCRY / 128, rd[i + 1] / 64), 256, 0, stream>>>(
        x, wRs[i], nullptr, rtmp, ping[i & 1], NCRY, rd[i + 1], rd[i], rd[i], 0);
    x = ping[i & 1];
  }
  out_head<<<(NCRY * 2 + 255) / 256, 256, 0, stream>>>(x, outW, outb, out, NCRY);
}
